// TreeGAT_69810398429647
// MI455X (gfx1250) — compile-verified, matrix operand order checked by reading, not run on hardware
//
#include <hip/hip_runtime.h>
#include <hip/hip_fp16.h>

typedef __attribute__((ext_vector_type(16))) _Float16 v16h;
typedef __attribute__((ext_vector_type(8)))  float    v8f;
typedef __attribute__((ext_vector_type(4)))  unsigned int v4u;
typedef __attribute__((ext_vector_type(8)))  int      v8i_;
typedef __attribute__((ext_vector_type(4)))  int      v4i_;

#define NND   50000      // nodes
#define SEQL  16
#define HID   100
#define NROOT 64
#define NE    100000
#define NEL   (NE + NND) // edges incl self loops

#define CDIV(a,b) (((a)+(b)-1)/(b))

// ---------------------------------------------------------------------------
// WMMA helpers (CDNA5 16x16x32 f16 -> f32)
// ---------------------------------------------------------------------------
static __device__ __forceinline__ v8f wmma_f16(v16h a, v16h b, v8f c) {
  return __builtin_amdgcn_wmma_f32_16x16x32_f16(false, a, false, b, (short)0, c, false, false);
}

// Low 32 bits of a generic LDS pointer == LDS byte address (aperture truncation).
static __device__ __forceinline__ unsigned lds_addr_of(const void* p) {
  return (unsigned)(size_t)p;
}

// A operand (16x32 f16) from a plain row-major f16 buffer (stride in halves).
// CDNA5 16-bit A layout: lanes 0-15 hold M=0..15 with K pairs (k0..k0+7, k0+16..k0+23),
// lanes 16-31 hold K (k0+8..k0+15, k0+24..k0+31).
static __device__ __forceinline__ v16h load_a16(const _Float16* s, int stride,
                                                int mbase, int k0, int lane) {
  int m  = mbase + (lane & 15);
  int kb = k0 + ((lane >> 4) << 3);
  const _Float16* p = s + (size_t)m * stride + kb;
  union { v16h v; uint4 q[2]; } u;
  u.q[0] = *(const uint4*)p;
  u.q[1] = *(const uint4*)(p + 16);
  return u.v;
}

// B operand from a pre-packed 512-half block: lane reads its 16 halves contiguously.
static __device__ __forceinline__ v16h load_b16(const _Float16* blk, int lane) {
  const uint4* p = (const uint4*)blk + lane * 2;
  union { v16h v; uint4 q[2]; } u;
  u.q[0] = p[0];
  u.q[1] = p[1];
  return u.v;
}

// Packed-B element mapping (mirror of A layout, lane = N, halves = K):
static __device__ __forceinline__ int b_klocal(int lane, int e) {
  return e + ((e >= 8) ? 8 : 0) + ((lane >> 4) << 3);
}

static __device__ __forceinline__ float sigmoidf_(float x) {
  return 1.0f / (1.0f + __expf(-x));
}
static __device__ __forceinline__ float leakyf_(float x) {
  return x > 0.0f ? x : 0.2f * x;
}

// ---------------------------------------------------------------------------
// Tensor Data Mover: DMA a 2D f16 tile [rows=64 x cols=tile_w] (row stride
// stride_elems, in elements) from global to LDS (packed, LDS row stride =
// tile_w). Rows >= rows_valid read as zero (TDM OOB semantics). One TDM op
// per calling wave; caller waits TENSORcnt and barriers.
// D# layout per CDNA5 ISA ch.8: group0 = {flags, lds_addr, global_addr, type},
// group1 = {wg_mask/data_size, dims/tiles/strides}.
// ---------------------------------------------------------------------------
static __device__ __forceinline__ void tdm_load_tile_f16(
    unsigned lds_addr, const _Float16* gptr, int tile_w, int rows_valid,
    int stride_elems) {
  unsigned long long ga = (unsigned long long)(size_t)gptr;
  v4u g0;
  g0[0] = 1u;                                          // count=1, user descriptor
  g0[1] = lds_addr;                                    // lds_addr [63:32]
  g0[2] = (unsigned)(ga & 0xffffffffu);                // global_addr lo
  g0[3] = (unsigned)((ga >> 32) & 0x01ffffffu) | (2u << 30); // addr hi | type=2
  v8i_ g1;
  g1[0] = 1 << 16;                                     // data_size=1 (2 bytes)
  g1[1] = (tile_w & 0xffff) << 16;                     // tensor_dim0 lo16 @ [63:48]
  g1[2] = ((tile_w >> 16) & 0xffff) |                  // tensor_dim0 hi16
          ((rows_valid & 0xffff) << 16);               // tensor_dim1 lo16
  g1[3] = ((rows_valid >> 16) & 0xffff) |              // tensor_dim1 hi16
          ((tile_w & 0xffff) << 16);                   // tile_dim0
  g1[4] = 64;                                          // tile_dim1=64, tile_dim2=0
  g1[5] = stride_elems;                                // tensor_dim0_stride lo32
  g1[6] = 0;                                           // stride hi, dim1_stride lo
  g1[7] = 0;
  v4i_ z = {0, 0, 0, 0};
#if defined(__clang_major__) && (__clang_major__ >= 23)
  v8i_ z8 = {0, 0, 0, 0, 0, 0, 0, 0};
  __builtin_amdgcn_tensor_load_to_lds(g0, g1, z, z, z8, 0);
#else
  __builtin_amdgcn_tensor_load_to_lds(g0, g1, z, z, 0);
#endif
}

// Per-lane async copy: 16 bytes global -> LDS, tracked by ASYNCcnt.
static __device__ __forceinline__ void async_g2l_b128(unsigned lds_addr,
                                                      const void* gptr) {
  unsigned long long ga = (unsigned long long)(size_t)gptr;
  asm volatile("global_load_async_to_lds_b128 %0, %1, off"
               :: "v"(lds_addr), "v"(ga) : "memory");
}
static __device__ __forceinline__ void wait_asynccnt0() {
  asm volatile("s_wait_asynccnt 0" ::: "memory");
}

// ---------------------------------------------------------------------------
// Packing kernels
// ---------------------------------------------------------------------------
__global__ void pack_emb_k(const float* __restrict__ emb, _Float16* __restrict__ o) {
  int i = blockIdx.x * 256 + threadIdx.x;
  if (i >= NND * 112) return;
  int n = i / 112, c = i % 112;
  o[i] = (_Float16)(c < HID ? emb[(size_t)n * HID + c] : 0.0f);
}

// GRU packed weights: K=256 ([x(0..127) | h(128..255)]), N=336 = 3 gates (r,z,n)
// each padded 100->112. Stored as [8 ktiles][21 ntiles] blocks of 512 halves.
__global__ void pack_gru_k(const float* __restrict__ Wih, const float* __restrict__ Whh,
                           _Float16* __restrict__ o) {
  int i = blockIdx.x * 256 + threadIdx.x;
  if (i >= 8 * 21 * 512) return;
  int blk = i >> 9, e16 = i & 511;
  int ks = blk / 21, nt = blk % 21;
  int lane = e16 >> 4, e = e16 & 15;
  int k = ks * 32 + b_klocal(lane, e);
  int g = nt / 7, jb = nt % 7;
  int j = jb * 16 + (lane & 15);
  float v = 0.0f;
  if (j < HID) {
    int row = g * HID + j;
    if (k < 128) { if (k < HID)       v = Wih[(size_t)row * HID + k]; }
    else         { int kk = k - 128; if (kk < HID) v = Whh[(size_t)row * HID + kk]; }
  }
  o[i] = (_Float16)v;
}

__global__ void pack_b_k(const float* __restrict__ W, int Krows, int Ncols,
                         int ksteps, int ntTotal, _Float16* __restrict__ o) {
  int i = blockIdx.x * 256 + threadIdx.x;
  int total = ksteps * ntTotal * 512;
  if (i >= total) return;
  int blk = i >> 9, e16 = i & 511;
  int ks = blk / ntTotal, nt = blk % ntTotal;
  int lane = e16 >> 4, e = e16 & 15;
  int k = ks * 32 + b_klocal(lane, e);
  int n = nt * 16 + (lane & 15);
  float v = (k < Krows && n < Ncols) ? W[(size_t)k * Ncols + n] : 0.0f;
  o[i] = (_Float16)v;
}

// ---------------------------------------------------------------------------
// Fused 2-layer GRU, 16 steps, 64 nodes / block, all GEMMs via WMMA
// ---------------------------------------------------------------------------
__device__ __forceinline__ void gru_layer(const _Float16* sX, _Float16* sH,
                                          const _Float16* Wp,
                                          const float* bih, const float* bhh,
                                          int lane, int wave) {
  float hnew[4][8];
#pragma unroll
  for (int s = 0; s < 4; ++s) {
    int tix = wave + s * 8;
    if (tix >= 28) break;                 // 4 Mtiles x 7 Ntiles
    int mt = tix / 7, jb = tix % 7;
    v8f cr = {}, cz = {}, cin = {}, chn = {};
#pragma unroll
    for (int ks = 0; ks < 4; ++ks) {      // x part: K 0..127
      v16h a  = load_a16(sX, 128, mt * 16, ks * 32, lane);
      v16h br = load_b16(Wp + ((size_t)(ks * 21 + jb)      << 9), lane);
      cr = wmma_f16(a, br, cr);
      v16h bz = load_b16(Wp + ((size_t)(ks * 21 + 7 + jb)  << 9), lane);
      cz = wmma_f16(a, bz, cz);
      v16h bn = load_b16(Wp + ((size_t)(ks * 21 + 14 + jb) << 9), lane);
      cin = wmma_f16(a, bn, cin);
    }
#pragma unroll
    for (int ks = 4; ks < 8; ++ks) {      // h part: K 128..255
      v16h a  = load_a16(sH, 128, mt * 16, (ks - 4) * 32, lane);
      v16h br = load_b16(Wp + ((size_t)(ks * 21 + jb)      << 9), lane);
      cr = wmma_f16(a, br, cr);
      v16h bz = load_b16(Wp + ((size_t)(ks * 21 + 7 + jb)  << 9), lane);
      cz = wmma_f16(a, bz, cz);
      v16h bh = load_b16(Wp + ((size_t)(ks * 21 + 14 + jb) << 9), lane);
      chn = wmma_f16(a, bh, chn);
    }
    int j = jb * 16 + (lane & 15);        // j < 112 < 128, in-bounds
    bool jv = j < HID;
    float br_ = jv ? bih[j] + bhh[j] : 0.0f;
    float bz_ = jv ? bih[HID + j] + bhh[HID + j] : 0.0f;
    float bi_ = jv ? bih[2 * HID + j] : 0.0f;
    float bh_ = jv ? bhh[2 * HID + j] : 0.0f;
    int m0 = mt * 16 + ((lane >> 4) << 3);
#pragma unroll
    for (int i = 0; i < 8; ++i) {
      float r = sigmoidf_(cr[i] + br_);
      float z = sigmoidf_(cz[i] + bz_);
      float n = tanhf(cin[i] + bi_ + r * (chn[i] + bh_));
      float ho = (float)sH[(size_t)(m0 + i) * 128 + j];
      hnew[s][i] = (1.0f - z) * n + z * ho;
    }
  }
  __syncthreads();                        // all reads of sH done
#pragma unroll
  for (int s = 0; s < 4; ++s) {
    int tix = wave + s * 8;
    if (tix >= 28) break;
    int mt = tix / 7, jb = tix % 7;
    int j = jb * 16 + (lane & 15);
    if (j < HID) {
      int m0 = mt * 16 + ((lane >> 4) << 3);
#pragma unroll
      for (int i = 0; i < 8; ++i)
        sH[(size_t)(m0 + i) * 128 + j] = (_Float16)hnew[s][i];
    }
  }
  __syncthreads();
}

__global__ __launch_bounds__(256) void gru_kernel(
    const int* __restrict__ tokens, const _Float16* __restrict__ embF16,
    const float* __restrict__ h0,
    const _Float16* __restrict__ Wp0, const _Float16* __restrict__ Wp1,
    const float* __restrict__ bih0, const float* __restrict__ bhh0,
    const float* __restrict__ bih1, const float* __restrict__ bhh1,
    _Float16* __restrict__ hlast) {
  __shared__ _Float16 sX[64 * 128];
  __shared__ _Float16 sH0[64 * 128];
  __shared__ _Float16 sH1[64 * 128];
  int tid = threadIdx.x;
  int lane = tid & 31, wave = tid >> 5;
  int mBase = blockIdx.x * 64;

  for (int i = tid; i < 64 * 128; i += 256) {
    int r = i >> 7, c = i & 127;
    int node = mBase + r;
    float v0 = 0.0f, v1 = 0.0f;
    if (node < NND && c < HID) {
      v0 = h0[(size_t)node * HID + c];
      v1 = h0[(size_t)(NND + node) * HID + c];
    }
    sH0[i] = (_Float16)v0;
    sH1[i] = (_Float16)v1;
  }
  // zero-pad x columns 112..127 once (gather never touches them)
  if (tid < 128) {
    int r = tid >> 1, c4 = 14 + (tid & 1);
    *(uint4*)&sX[(size_t)r * 128 + c4 * 8] = make_uint4(0, 0, 0, 0);
  }
  __syncthreads();

  for (int t = 0; t < SEQL; ++t) {
    // token-indexed gather: per-lane async global->LDS copies (ASYNCcnt)
    for (int i = tid; i < 64 * 14; i += 256) {
      int r = i / 14, c4 = i % 14;
      int node = mBase + r;
      int tok = (node < NND) ? tokens[(size_t)node * SEQL + t] : 0;
      async_g2l_b128(lds_addr_of(&sX[(size_t)r * 128 + c4 * 8]),
                     &embF16[(size_t)tok * 112 + c4 * 8]);
    }
    wait_asynccnt0();
    __syncthreads();
    gru_layer(sX,  sH0, Wp0, bih0, bhh0, lane, wave);   // layer0: x=emb
    gru_layer(sH0, sH1, Wp1, bih1, bhh1, lane, wave);   // layer1: x=h0
  }

  for (int i = tid; i < 64 * 128; i += 256) {
    int r = i >> 7, c = i & 127;
    int node = mBase + r;
    if (node < NND) hlast[(size_t)node * 128 + c] = sH1[(size_t)r * 128 + c];
  }
}

__global__ void root_replace_k(const float* __restrict__ ure, _Float16* __restrict__ hlast) {
  int i = blockIdx.x * 256 + threadIdx.x;
  if (i >= NROOT * 128) return;
  int n = i >> 7, c = i & 127;
  hlast[i] = (_Float16)(c < HID ? ure[(size_t)n * HID + c] : 0.0f);
}

// ---------------------------------------------------------------------------
// Generic f16 WMMA GEMM: Out[M x Ncols](f16) = A[M x K](f16) * Bpacked.
// A tiles are DMAed into LDS by the Tensor Data Mover (TENSORcnt).
// ---------------------------------------------------------------------------
__global__ __launch_bounds__(256) void gemm_f16_k(
    const _Float16* __restrict__ A, int Astride, int M, int K,
    const _Float16* __restrict__ Bblk, int ntTotal,
    _Float16* __restrict__ Out, int Ostride, int Ncols) {
  __shared__ _Float16 sA[64 * 128];
  int tid = threadIdx.x, lane = tid & 31, wave = tid >> 5;
  int mBase = blockIdx.x * 64;
  int ntBase = blockIdx.y * 7;
  int ntCount = ntTotal - ntBase;
  if (ntCount > 7) ntCount = 7;
  int nTiles = 4 * ntCount;
  int rowsValid = M - mBase;   // TDM returns zero for rows >= this

  v8f acc[4] = {};
  for (int kc = 0; kc < K; kc += 128) {
    int chunk = K - kc; if (chunk > 128) chunk = 128;
    if (wave == 0) {
      tdm_load_tile_f16(lds_addr_of(&sA[0]),
                        A + (size_t)mBase * Astride + kc,
                        chunk, rowsValid, Astride);
      __builtin_amdgcn_s_wait_tensorcnt(0);
    }
    __syncthreads();
    int ksteps = chunk >> 5;
#pragma unroll
    for (int s = 0; s < 4; ++s) {
      int tix = wave + s * 8;
      if (tix >= nTiles) break;
      int mt = tix / ntCount, ntG = ntBase + tix % ntCount;
      for (int ks = 0; ks < ksteps; ++ks) {
        v16h a = load_a16(sA, chunk, mt * 16, ks * 32, lane);
        v16h b = load_b16(Bblk + ((size_t)(((kc >> 5) + ks) * ntTotal + ntG) << 9), lane);
        acc[s] = wmma_f16(a, b, acc[s]);
      }
    }
    __syncthreads();
  }
#pragma unroll
  for (int s = 0; s < 4; ++s) {
    int tix = wave + s * 8;
    if (tix >= nTiles) break;
    int mt = tix / ntCount, ntG = ntBase + tix % ntCount;
    int n = ntG * 16 + (lane & 15);
    if (n < Ncols) {
      int m0 = mBase + mt * 16 + ((lane >> 4) << 3);
#pragma unroll
      for (int i = 0; i < 8; ++i) {
        int row = m0 + i;
        if (row < M) Out[(size_t)row * Ostride + n] = (_Float16)acc[s][i];
      }
    }
  }
}

// ---------------------------------------------------------------------------
// GAT attention pieces
// ---------------------------------------------------------------------------
__global__ void attn_coef_k(const _Float16* __restrict__ h, int stride, int heads, int D,
                            const float* __restrict__ as_w, const float* __restrict__ ad_w,
                            float* __restrict__ as_o, float* __restrict__ ad_o) {
  int i = blockIdx.x * 256 + threadIdx.x;
  if (i >= NND * heads) return;
  int n = i / heads, hh = i % heads;
  const _Float16* row = h + (size_t)n * stride + hh * D;
  float s = 0.0f, d = 0.0f;
  for (int c = 0; c < D; ++c) {
    float v = (float)row[c];
    s += v * as_w[hh * D + c];
    d += v * ad_w[hh * D + c];
  }
  as_o[i] = s;
  ad_o[i] = d;
}

__global__ void fill_f32_k(float* __restrict__ p, float v, size_t n) {
  size_t i = (size_t)blockIdx.x * 256 + threadIdx.x;
  if (i < n) p[i] = v;
}

static __device__ __forceinline__ void edge_nodes(const int* ei, int e, int& s, int& d) {
  if (e < NE) { s = ei[e]; d = ei[NE + e]; }
  else        { s = d = e - NE; }
}

static __device__ __forceinline__ void atomic_max_f32(float* addr, float val) {
  if (!(val < 0.0f)) atomicMax((int*)addr, __float_as_int(val));
  else               atomicMin((unsigned int*)addr, __float_as_uint(val));
}

__global__ void edge_max_k(const int* __restrict__ ei, const float* __restrict__ as_v,
                           const float* __restrict__ ad_v, float* __restrict__ m, int heads) {
  int i = blockIdx.x * 256 + threadIdx.x;
  if (i >= NEL * heads) return;
  int e = i / heads, hh = i % heads;
  int s, d; edge_nodes(ei, e, s, d);
  float x = leakyf_(as_v[(size_t)s * heads + hh] + ad_v[(size_t)d * heads + hh]);
  atomic_max_f32(&m[(size_t)d * heads + hh], x);
}

__global__ void edge_exp_k(const int* __restrict__ ei, const float* __restrict__ as_v,
                           const float* __restrict__ ad_v, const float* __restrict__ m,
                           float* __restrict__ ex, float* __restrict__ den, int heads) {
  int i = blockIdx.x * 256 + threadIdx.x;
  if (i >= NEL * heads) return;
  int e = i / heads, hh = i % heads;
  int s, d; edge_nodes(ei, e, s, d);
  float x = leakyf_(as_v[(size_t)s * heads + hh] + ad_v[(size_t)d * heads + hh]);
  float v = __expf(x - m[(size_t)d * heads + hh]);
  ex[i] = v;
  unsafeAtomicAdd(&den[(size_t)d * heads + hh], v);
}

__global__ void edge_agg_k(const int* __restrict__ ei, const float* __restrict__ ex,
                           const float* __restrict__ den, const _Float16* __restrict__ h,
                           int stride, int heads, int D, float* __restrict__ outp, int ostride) {
  int i = blockIdx.x * 256 + threadIdx.x;
  if (i >= NEL * heads) return;
  int e = i / heads, hh = i % heads;
  int s, d; edge_nodes(ei, e, s, d);
  float alpha = ex[i] / (den[(size_t)d * heads + hh] + 1e-16f);
  const _Float16* hrow = h + (size_t)s * stride + hh * D;
  float* orow = outp + (size_t)d * ostride + hh * D;
  for (int c = 0; c < D; ++c)
    unsafeAtomicAdd(&orow[c], (float)hrow[c] * alpha);
}

__global__ void bias_cvt_f16_k(const float* __restrict__ acc, const float* __restrict__ bias,
                               int cols, _Float16* __restrict__ o, size_t n) {
  size_t i = (size_t)blockIdx.x * 256 + threadIdx.x;
  if (i < n) o[i] = (_Float16)(acc[i] + bias[i % cols]);
}

__global__ void bias_add_k(float* __restrict__ o, const float* __restrict__ bias,
                           int cols, size_t n) {
  size_t i = (size_t)blockIdx.x * 256 + threadIdx.x;
  if (i < n) o[i] += bias[i % cols];
}

// ---------------------------------------------------------------------------
extern "C" void kernel_launch(void* const* d_in, const int* in_sizes, int n_in,
                              void* d_out, int out_size, void* d_ws, size_t ws_size,
                              hipStream_t stream) {
  (void)in_sizes; (void)n_in; (void)out_size; (void)ws_size;
  const int*   tokens = (const int*)d_in[0];
  const int*   ei     = (const int*)d_in[1];
  const float* ure    = (const float*)d_in[2];
  const float* emb    = (const float*)d_in[3];
  const float* h0     = (const float*)d_in[4];
  const float* Wih0   = (const float*)d_in[5];
  const float* Whh0   = (const float*)d_in[6];
  const float* bih0   = (const float*)d_in[7];
  const float* bhh0   = (const float*)d_in[8];
  const float* Wih1   = (const float*)d_in[9];
  const float* Whh1   = (const float*)d_in[10];
  const float* bih1   = (const float*)d_in[11];
  const float* bhh1   = (const float*)d_in[12];
  const float* W1     = (const float*)d_in[13];
  const float* atts1  = (const float*)d_in[14];
  const float* attd1  = (const float*)d_in[15];
  const float* bias1  = (const float*)d_in[16];
  const float* W2     = (const float*)d_in[17];
  const float* atts2  = (const float*)d_in[18];
  const float* attd2  = (const float*)d_in[19];
  const float* bias2  = (const float*)d_in[20];
  float* out = (float*)d_out;

  char* w = (char*)d_ws;
  size_t off = 0;
  auto alloc = [&](size_t bytes) -> char* {
    char* p = w + off;
    off = (off + bytes + 255) & ~(size_t)255;
    return p;
  };
  _Float16* embF16 = (_Float16*)alloc((size_t)NND * 112 * 2);  // reused as h2f16 later
  _Float16* Wp0    = (_Float16*)alloc(8 * 21 * 512 * 2);
  _Float16* Wp1    = (_Float16*)alloc(8 * 21 * 512 * 2);
  _Float16* W1p    = (_Float16*)alloc(4 * 50 * 512 * 2);
  _Float16* W2p    = (_Float16*)alloc(25 * 7 * 512 * 2);
  _Float16* hlast  = (_Float16*)alloc((size_t)NND * 128 * 2);
  _Float16* h1f16  = (_Float16*)alloc((size_t)NND * 800 * 2); // reused as y1f16 after agg
  float*    y1agg  = (float*)   alloc((size_t)NND * 800 * 4);
  float*    as1    = (float*)   alloc((size_t)NND * 8 * 4);
  float*    ad1    = (float*)   alloc((size_t)NND * 8 * 4);
  float*    m1     = (float*)   alloc((size_t)NND * 8 * 4);
  float*    den1   = (float*)   alloc((size_t)NND * 8 * 4);
  float*    ex1    = (float*)   alloc((size_t)NEL * 8 * 4);
  float*    as2    = (float*)   alloc((size_t)NND * 4);
  float*    ad2    = (float*)   alloc((size_t)NND * 4);
  float*    m2     = (float*)   alloc((size_t)NND * 4);
  float*    den2   = (float*)   alloc((size_t)NND * 4);
  float*    ex2    = (float*)   alloc((size_t)NEL * 4);

  const float NEGINF = -__builtin_inff();

  // ---- pack f16 operands --------------------------------------------------
  pack_emb_k<<<CDIV(NND * 112, 256), 256, 0, stream>>>(emb, embF16);
  pack_gru_k<<<CDIV(8 * 21 * 512, 256), 256, 0, stream>>>(Wih0, Whh0, Wp0);
  pack_gru_k<<<CDIV(8 * 21 * 512, 256), 256, 0, stream>>>(Wih1, Whh1, Wp1);
  pack_b_k<<<CDIV(4 * 50 * 512, 256), 256, 0, stream>>>(W1, 100, 800, 4, 50, W1p);
  pack_b_k<<<CDIV(25 * 7 * 512, 256), 256, 0, stream>>>(W2, 800, 100, 25, 7, W2p);

  // ---- fused 2-layer GRU --------------------------------------------------
  gru_kernel<<<CDIV(NND, 64), 256, 0, stream>>>(tokens, embF16, h0, Wp0, Wp1,
                                                bih0, bhh0, bih1, bhh1, hlast);
  root_replace_k<<<CDIV(NROOT * 128, 256), 256, 0, stream>>>(ure, hlast);

  // ---- GAT layer 1 --------------------------------------------------------
  {
    dim3 g(CDIV(NND, 64), 8);  // 50 N-tiles in slabs of 7
    gemm_f16_k<<<g, 256, 0, stream>>>(hlast, 128, NND, 128, W1p, 50, h1f16, 800, 800);
  }
  attn_coef_k<<<CDIV(NND * 8, 256), 256, 0, stream>>>(h1f16, 800, 8, 100,
                                                      atts1, attd1, as1, ad1);
  fill_f32_k<<<CDIV(NND * 8, 256), 256, 0, stream>>>(m1, NEGINF, (size_t)NND * 8);
  fill_f32_k<<<CDIV(NND * 8, 256), 256, 0, stream>>>(den1, 0.0f, (size_t)NND * 8);
  fill_f32_k<<<CDIV(NND * 800, 256), 256, 0, stream>>>(y1agg, 0.0f, (size_t)NND * 800);
  edge_max_k<<<CDIV(NEL * 8, 256), 256, 0, stream>>>(ei, as1, ad1, m1, 8);
  edge_exp_k<<<CDIV(NEL * 8, 256), 256, 0, stream>>>(ei, as1, ad1, m1, ex1, den1, 8);
  edge_agg_k<<<CDIV(NEL * 8, 256), 256, 0, stream>>>(ei, ex1, den1, h1f16, 800, 8, 100,
                                                     y1agg, 800);
  bias_cvt_f16_k<<<CDIV(NND * 800, 256), 256, 0, stream>>>(y1agg, bias1, 800, h1f16,
                                                           (size_t)NND * 800);

  // ---- GAT layer 2 (1 head) ----------------------------------------------
  _Float16* h2f16 = embF16;  // reuse embedding buffer ([NND][112])
  {
    dim3 g(CDIV(NND, 64), 1);  // 7 N-tiles
    gemm_f16_k<<<g, 256, 0, stream>>>(h1f16, 800, NND, 800, W2p, 7, h2f16, 112, 112);
  }
  attn_coef_k<<<CDIV(NND, 256), 256, 0, stream>>>(h2f16, 112, 1, 100,
                                                  atts2, attd2, as2, ad2);
  fill_f32_k<<<CDIV(NND, 256), 256, 0, stream>>>(m2, NEGINF, (size_t)NND);
  fill_f32_k<<<CDIV(NND, 256), 256, 0, stream>>>(den2, 0.0f, (size_t)NND);
  fill_f32_k<<<CDIV(NND * 100, 256), 256, 0, stream>>>(out, 0.0f, (size_t)NND * 100);
  edge_max_k<<<CDIV(NEL, 256), 256, 0, stream>>>(ei, as2, ad2, m2, 1);
  edge_exp_k<<<CDIV(NEL, 256), 256, 0, stream>>>(ei, as2, ad2, m2, ex2, den2, 1);
  edge_agg_k<<<CDIV(NEL, 256), 256, 0, stream>>>(ei, ex2, den2, h2f16, 112, 1, 100,
                                                 out, 100);
  bias_add_k<<<CDIV(NND * 100, 256), 256, 0, stream>>>(out, bias2, 100, (size_t)NND * 100);
}
